// CascadeSystem_26319559590184
// MI455X (gfx1250) — matrix operand order repaired and checked by reading, not compiled, not run on hardware
//
#include <hip/hip_runtime.h>

// ---------------------------------------------------------------------------
// CascadeSystem on MI455X (gfx1250, wave32).
//   projected = input @ W^T + b        (f32 WMMA 16x16x4, exact, runs once)
//   split: conn = connHi + connLo      (f16 hi/lo pair, ~22 mantissa bits)
//   100x:  active = state > 1 (f16 {0,1}, exact) ; done/total bookkeeping
//          prop   = active @ (connHi + connLo)^T   (f16 WMMA 16x16x32,
//                   f32 accumulate)
//          state  = done ? state : (active ? 0 : 0.9*state + prop)
//   out = [state (B*N), max(total) (1)]
//
// Propagation kernel: one block per 64-column N-group; its 8 waves cover all
// 8 M-tiles (BATCH=128) and share the B tile (conn hi/lo) through LDS with
// double buffering. conn is then read from L2 exactly once per iteration
// (64 MB) instead of 8x (512 MB) -> matrix-pipe bound, not L2 bound.
// ---------------------------------------------------------------------------

typedef __attribute__((ext_vector_type(2)))  float        v2f;
typedef __attribute__((ext_vector_type(8)))  float        v8f;
typedef __attribute__((ext_vector_type(16))) _Float16     v16h;
typedef __attribute__((ext_vector_type(4)))  unsigned int v4u;   // 16 bytes

#define NODES 4096
#define BATCH 128
#define NT_PER_WAVE 4              // 4 x 16 = 64 columns per wave / block
#define KC 64                      // K-chunk staged in LDS per buffer
#define LDS_STRIDE 72              // 64 + 8 f16 pad -> conflict-light b128 reads
#define MAX_ITERS 100

union F16FragA { v16h h; v4u u[2]; };

// ------------------------- one-time conn -> hi/lo split ---------------------
__global__ __launch_bounds__(256) void cascade_split_kernel(
    const float* __restrict__ conn,
    _Float16* __restrict__ hi,
    _Float16* __restrict__ lo)
{
    const size_t total = (size_t)NODES * NODES;
    const size_t stride = (size_t)gridDim.x * blockDim.x;
    for (size_t i = blockIdx.x * (size_t)blockDim.x + threadIdx.x; i < total; i += stride) {
        const float c = conn[i];
        const _Float16 h = (_Float16)c;
        hi[i] = h;
        lo[i] = (_Float16)(c - (float)h);
    }
}

// ----------------- projection GEMM: state = input @ W^T + b -----------------
// Runs once; exact f32 via V_WMMA_F32_16X16X4_F32.
__global__ __launch_bounds__(256) void cascade_proj_kernel(
    const float* __restrict__ A,        // [BATCH x NODES]
    const float* __restrict__ W,        // [NODES x NODES]
    const float* __restrict__ bias,     // [NODES]
    float* __restrict__ state)          // [BATCH x NODES]
{
    const int wave = (int)((blockIdx.x * blockDim.x + threadIdx.x) >> 5);
    const int lane = (int)(threadIdx.x & 31);
    const int nGroups = NODES / (16 * NT_PER_WAVE);       // 64
    const int m0 = (wave / nGroups) * 16;
    const int n0 = (wave % nGroups) * (16 * NT_PER_WAVE);
    const int half = lane >> 4;
    const int l15  = lane & 15;

    v8f acc[NT_PER_WAVE] = {};
    const float* aPtr  = A + (size_t)(m0 + l15) * NODES + half * 2;
    const float* bBase = W + (size_t)(n0 + l15) * NODES + half * 2;

    for (int k0 = 0; k0 < NODES; k0 += 4) {
        v2f afrag = *(const v2f*)(aPtr + k0);
#pragma unroll
        for (int t = 0; t < NT_PER_WAVE; ++t) {
            v2f bfrag = *(const v2f*)(bBase + (size_t)t * 16 * NODES + k0);
            acc[t] = __builtin_amdgcn_wmma_f32_16x16x4_f32(
                false, afrag, false, bfrag, (short)0, acc[t], false, false);
        }
    }
#pragma unroll
    for (int t = 0; t < NT_PER_WAVE; ++t) {
        const int n = n0 + t * 16 + l15;
        const float bv = bias[n];
#pragma unroll
        for (int r = 0; r < 8; ++r) {
            const int m = m0 + r + half * 8;
            state[(size_t)m * NODES + n] = acc[t][r] + bv;
        }
    }
}

// ------------- propagation GEMM (f16 split, LDS-staged B) + update ----------
// Block b owns columns [b*64, b*64+64); wave w owns rows [w*16, w*16+16).
// prop[m,n] = sum_k active[m,k] * (connHi[n,k] + connLo[n,k]), f32 accumulate.
__global__ __launch_bounds__(256) void cascade_prop_kernel(
    const _Float16* __restrict__ activeH,  // [BATCH x NODES], values {0,1}
    const _Float16* __restrict__ connHi,   // [NODES x NODES]
    const _Float16* __restrict__ connLo,   // [NODES x NODES]
    float* __restrict__ state,             // [BATCH x NODES] in-out
    const int* __restrict__ done)          // [BATCH]
{
    __shared__ _Float16 ldsB[2][2][64][LDS_STRIDE];   // [buf][hi/lo][row][k] 36 KB

    const int tid  = (int)threadIdx.x;
    const int wave = tid >> 5;          // 0..7  -> M-tile
    const int lane = tid & 31;
    const int half = lane >> 4;
    const int l15  = lane & 15;
    const int m0   = wave * 16;
    const int n0   = (int)blockIdx.x * 64;

    // Cooperative B staging: 64 rows x KC(=64) f16 per matrix = 512 b128
    // units; 256 threads handle 2 units each (per matrix).
    const int u0 = tid,        r0 = u0 >> 3, c0 = (u0 & 7) * 8;
    const int u1 = tid + 256,  r1 = u1 >> 3, c1 = (u1 & 7) * 8;
    const _Float16* gHi0 = connHi + (size_t)(n0 + r0) * NODES + c0;
    const _Float16* gHi1 = connHi + (size_t)(n0 + r1) * NODES + c1;
    const _Float16* gLo0 = connLo + (size_t)(n0 + r0) * NODES + c0;
    const _Float16* gLo1 = connLo + (size_t)(n0 + r1) * NODES + c1;

    // Preload chunk 0 into buffer 0.
    {
        v4u h0 = *(const v4u*)gHi0;
        v4u h1 = *(const v4u*)gHi1;
        v4u l0 = *(const v4u*)gLo0;
        v4u l1 = *(const v4u*)gLo1;
        *(v4u*)&ldsB[0][0][r0][c0] = h0;
        *(v4u*)&ldsB[0][0][r1][c1] = h1;
        *(v4u*)&ldsB[0][1][r0][c0] = l0;
        *(v4u*)&ldsB[0][1][r1][c1] = l1;
    }
    __syncthreads();

    v8f acc[NT_PER_WAVE] = {};
    const _Float16* aRow = activeH + (size_t)(m0 + l15) * NODES + half * 16;

    const int NCHUNK = NODES / KC;   // 64
    for (int c = 0; c < NCHUNK; ++c) {
        const int k0  = c * KC;
        const int buf = c & 1;
        const bool more = (c + 1) < NCHUNK;

        // Prefetch next chunk global -> registers (overlaps WMMA below).
        v4u nh0 = {}, nh1 = {}, nl0 = {}, nl1 = {};
        if (more) {
            const int kn = k0 + KC;
            nh0 = *(const v4u*)(gHi0 + kn);
            nh1 = *(const v4u*)(gHi1 + kn);
            nl0 = *(const v4u*)(gLo0 + kn);
            nl1 = *(const v4u*)(gLo1 + kn);
        }

        // Compute on current buffer.
#pragma unroll
        for (int ks = 0; ks < KC; ks += 32) {
            F16FragA a;
            a.u[0] = *(const v4u*)(aRow + k0 + ks);        // K = +half*16 .. +7
            a.u[1] = *(const v4u*)(aRow + k0 + ks + 8);    // K = +half*16+8 .. +15
#pragma unroll
            for (int t = 0; t < NT_PER_WAVE; ++t) {
                const _Float16* ph = &ldsB[buf][0][t * 16 + l15][ks + half * 16];
                const _Float16* pl = &ldsB[buf][1][t * 16 + l15][ks + half * 16];
                v16h bh = *(const v16h*)ph;
                v16h bl = *(const v16h*)pl;
                acc[t] = __builtin_amdgcn_wmma_f32_16x16x32_f16(
                    false, a.h, false, bh, (short)0, acc[t], false, false);
                acc[t] = __builtin_amdgcn_wmma_f32_16x16x32_f16(
                    false, a.h, false, bl, (short)0, acc[t], false, false);
            }
        }

        __syncthreads();
        if (more) {
            *(v4u*)&ldsB[buf ^ 1][0][r0][c0] = nh0;
            *(v4u*)&ldsB[buf ^ 1][0][r1][c1] = nh1;
            *(v4u*)&ldsB[buf ^ 1][1][r0][c0] = nl0;
            *(v4u*)&ldsB[buf ^ 1][1][r1][c1] = nl1;
            __syncthreads();
        }
    }

    // Epilogue: active is a pure function of state, recompute it here.
#pragma unroll
    for (int t = 0; t < NT_PER_WAVE; ++t) {
        const int n = n0 + t * 16 + l15;
#pragma unroll
        for (int r = 0; r < 8; ++r) {
            const int m = m0 + r + half * 8;
            const size_t idx = (size_t)m * NODES + n;
            const float s  = state[idx];
            const bool  av = (s > 1.0f);
            const float ns = av ? 0.0f : fmaf(0.9f, s, acc[t][r]);
            state[idx] = done[m] ? s : ns;
        }
    }
}

// --------- per-sample mask (f16 {0,1}) + n_active + done/total update -------
__global__ __launch_bounds__(256) void cascade_mask_kernel(
    const float* __restrict__ state,
    _Float16* __restrict__ activeH,
    int* __restrict__ done,
    float* __restrict__ total)
{
    const int b = (int)blockIdx.x;
    __shared__ float red[256];
    float cnt = 0.0f;
    for (int j = (int)threadIdx.x; j < NODES; j += 256) {
        const bool a = state[(size_t)b * NODES + j] > 1.0f;
        activeH[(size_t)b * NODES + j] = a ? (_Float16)1.0f : (_Float16)0.0f;
        cnt += a ? 1.0f : 0.0f;
    }
    red[threadIdx.x] = cnt;
    __syncthreads();
    for (int s = 128; s > 0; s >>= 1) {
        if ((int)threadIdx.x < s) red[threadIdx.x] += red[threadIdx.x + s];
        __syncthreads();
    }
    if (threadIdx.x == 0) {
        const float n_active = red[0];
        const int nd = done[b] | ((n_active == 0.0f) ? 1 : 0);
        total[b] += nd ? 0.0f : n_active;
        done[b] = nd;
    }
}

__global__ void cascade_init_kernel(int* __restrict__ done, float* __restrict__ total)
{
    const int i = (int)threadIdx.x;
    if (i < BATCH) { done[i] = 0; total[i] = 0.0f; }
}

__global__ void cascade_max_kernel(const float* __restrict__ total, float* __restrict__ out)
{
    __shared__ float red[BATCH];
    red[threadIdx.x] = total[threadIdx.x];
    __syncthreads();
    for (int s = BATCH / 2; s > 0; s >>= 1) {
        if ((int)threadIdx.x < s) red[threadIdx.x] = fmaxf(red[threadIdx.x], red[threadIdx.x + s]);
        __syncthreads();
    }
    if (threadIdx.x == 0) out[0] = red[0];
}

// ---------------------------------------------------------------------------
extern "C" void kernel_launch(void* const* d_in, const int* in_sizes, int n_in,
                              void* d_out, int out_size, void* d_ws, size_t ws_size,
                              hipStream_t stream) {
    (void)in_sizes; (void)n_in; (void)out_size; (void)ws_size;

    const float* input = (const float*)d_in[0];   // [BATCH, NODES]
    const float* W     = (const float*)d_in[1];   // [NODES, NODES]
    const float* bias  = (const float*)d_in[2];   // [NODES]
    const float* conn  = (const float*)d_in[3];   // [NODES, NODES]

    float* out    = (float*)d_out;
    float* state  = out;                          // first BATCH*NODES floats
    float* scalar = out + (size_t)BATCH * NODES;  // max avalanche size

    // Workspace: connHi | connLo | activeH | total | done  (~65 MB)
    _Float16* connHi  = (_Float16*)d_ws;
    _Float16* connLo  = connHi + (size_t)NODES * NODES;
    _Float16* activeH = connLo + (size_t)NODES * NODES;
    float*    total   = (float*)(activeH + (size_t)BATCH * NODES);
    int*      done    = (int*)(total + BATCH);

    const int projWaves  = (BATCH / 16) * (NODES / (16 * NT_PER_WAVE));  // 512
    const int projBlocks = projWaves * 32 / 256;                         // 64
    const int propBlocks = NODES / 64;                                   // 64

    cascade_init_kernel<<<1, BATCH, 0, stream>>>(done, total);

    // One-time f32 -> f16 hi/lo split of conn (amortized over 100 GEMMs).
    cascade_split_kernel<<<4096, 256, 0, stream>>>(conn, connHi, connLo);

    // Projection (exact f32 WMMA path).
    cascade_proj_kernel<<<projBlocks, 256, 0, stream>>>(input, W, bias, state);

    // Avalanche scan: fixed trip count matching lax.scan(length=100) with
    // per-sample sticky done; launch sequence is deterministic.
    for (int it = 0; it < MAX_ITERS; ++it) {
        cascade_mask_kernel<<<BATCH, 256, 0, stream>>>(state, activeH, done, total);
        cascade_prop_kernel<<<propBlocks, 256, 0, stream>>>(
            activeH, connHi, connLo, state, done);
    }

    cascade_max_kernel<<<1, BATCH, 0, stream>>>(total, scalar);
}
